// HighwayBiLSTM_40596030882337
// MI455X (gfx1250) — matrix operand order repaired and checked by reading, not compiled, not run on hardware
//
#include <hip/hip_runtime.h>

typedef __attribute__((ext_vector_type(16))) __bf16 v16bf_t;
typedef __attribute__((ext_vector_type(8)))  __bf16 v8bf_t;
typedef __attribute__((ext_vector_type(8)))  float  v8f_t;
typedef __attribute__((ext_vector_type(4)))  unsigned int v4u_t;
typedef __attribute__((ext_vector_type(8)))  unsigned int v8u_t;

#define TT   256
#define BB   64
#define HH   300
#define KP   320          // padded K (and padded gate width)
#define KTN  10           // K tiles of 32
#define NT6  120          // 6 gates * 320 / 16
#define NT5  100          // 5 gates * 320 / 16
#define NP6  1920         // padded gx width
#define ROWS (TT*BB)      // 16384
#define NWG  10           // persistent workgroups in scan

__device__ __forceinline__ float sigf(float x) { return 1.0f / (1.0f + __expf(-x)); }

__device__ __forceinline__ v8f_t wmma_bf16(v16bf_t a, v16bf_t b, v8f_t c) {
  return __builtin_amdgcn_wmma_f32_16x16x32_bf16(false, a, false, b, (short)0, c, false, false);
}

// A-fragment (16x32 bf16, row-major source, 320-elem row stride)
__device__ __forceinline__ v16bf_t load_a_frag(const __bf16* p) {
  v8bf_t lo = *(const v8bf_t*)p;        // K = kb .. kb+7
  v8bf_t hi = *(const v8bf_t*)(p + 16); // K = 16+kb .. 16+kb+7
  v16bf_t r;
#pragma unroll
  for (int i = 0; i < 8; ++i) { r[i] = lo[i]; r[i + 8] = hi[i]; }
  return r;
}

// B-fragment: 32 contiguous bytes per lane (global or LDS pointers)
__device__ __forceinline__ v16bf_t load_b_frag(const __bf16* p) {
  v8bf_t lo = *(const v8bf_t*)p;
  v8bf_t hi = *(const v8bf_t*)(p + 8);
  v16bf_t r;
#pragma unroll
  for (int i = 0; i < 8; ++i) { r[i] = lo[i]; r[i + 8] = hi[i]; }
  return r;
}

// ---- pack f32 [rows,300] -> bf16 padded row-major [rows,320] ----
__global__ __launch_bounds__(256) void pack_x_kernel(const float* __restrict__ src,
                                                     __bf16* __restrict__ dst, int n) {
  int idx = blockIdx.x * blockDim.x + threadIdx.x;
  if (idx >= n) return;
  int row = idx / KP, k = idx - row * KP;
  float v = (k < HH) ? src[row * HH + k] : 0.0f;
  dst[idx] = (__bf16)v;
}

// ---- pack f32 W[300, gates*300] -> bf16 B-fragment blocks [kt][ntG][lane][e] ----
__global__ __launch_bounds__(256) void pack_w_kernel(const float* __restrict__ W,
                                                     __bf16* __restrict__ dst,
                                                     int gates, int NT, int n) {
  int idx = blockIdx.x * blockDim.x + threadIdx.x;
  if (idx >= n) return;
  int e    = idx & 15;
  int lane = (idx >> 4) & 31;
  int rest = idx >> 9;
  int ntG  = rest % NT;
  int kt   = rest / NT;
  int k    = kt * 32 + ((lane & 16) ? 8 : 0) + (e & 7) + ((e & 8) ? 16 : 0);
  int colp = ntG * 16 + (lane & 15);
  int g    = colp / KP;
  int c    = colp - g * KP;
  float v  = 0.0f;
  if (k < HH && c < HH) v = W[k * (gates * HH) + g * HH + c];
  dst[idx] = (__bf16)v;
}

// ---- gx GEMM: [16384,320]x[320,1920] -> f32, 1x4 strips, double-buffered K ----
__global__ __launch_bounds__(256) void gemm_gx_kernel(const __bf16* __restrict__ Abf,
                                                      const __bf16* __restrict__ Wfrag,
                                                      float* __restrict__ gx) {
  int gw   = (blockIdx.x * 256 + threadIdx.x) >> 5; // 30720 wave jobs
  int lane = threadIdx.x & 31;
  int rt = gw / 30, strip = gw - rt * 30;
  v8f_t acc[4] = {v8f_t{}, v8f_t{}, v8f_t{}, v8f_t{}};
  int row = rt * 16 + (lane & 15);
  int kb  = (lane & 16) ? 8 : 0;

  v16bf_t a = load_a_frag(Abf + (size_t)row * KP + kb);
  v16bf_t b[4];
#pragma unroll
  for (int j = 0; j < 4; ++j)
    b[j] = load_b_frag(Wfrag + ((size_t)(strip * 4 + j) * 32 + lane) * 16);

#pragma unroll
  for (int kt = 0; kt < KTN; ++kt) {
    v16bf_t an;
    v16bf_t bn[4];
    if (kt < KTN - 1) {
      an = load_a_frag(Abf + (size_t)row * KP + (kt + 1) * 32 + kb);
#pragma unroll
      for (int j = 0; j < 4; ++j)
        bn[j] = load_b_frag(Wfrag + ((size_t)((kt + 1) * NT6 + strip * 4 + j) * 32 + lane) * 16);
    }
#pragma unroll
    for (int j = 0; j < 4; ++j) acc[j] = wmma_bf16(a, b[j], acc[j]);
    if (kt < KTN - 1) {
      a = an;
#pragma unroll
      for (int j = 0; j < 4; ++j) b[j] = bn[j];
    }
  }
  int r0 = (lane & 16) ? 8 : 0;
  int col = lane & 15;
#pragma unroll
  for (int j = 0; j < 4; ++j) {
    int nt = strip * 4 + j;
#pragma unroll
    for (int r = 0; r < 8; ++r)
      gx[(size_t)(rt * 16 + r + r0) * NP6 + nt * 16 + col] = acc[j][r];
  }
}

// ---- persistent recurrent scan: 10 WGs, Whh slice LDS-resident via TDM ----
__global__ __launch_bounds__(256, 1) void scan_kernel(const float* __restrict__ gx,
                                                      const __bf16* __restrict__ whhfrag,
                                                      const float* __restrict__ bias,
                                                      const float* __restrict__ masks,
                                                      float* __restrict__ seq_out,
                                                      float* __restrict__ hc_out,
                                                      __bf16* __restrict__ h_bf,
                                                      unsigned* __restrict__ bar,
                                                      int scanIdx, int rev) {
  // LDS: 100KB weight slice [kt][g][nt][lane][16] + 40KB gate exchange
  __shared__ __align__(16) __bf16 wlds[51200];
  __shared__ float gh[5 * BB * 32];
  const int tid = threadIdx.x, lane = tid & 31, wave = tid >> 5;
  const int wg = blockIdx.x;

  // ---- TDM preload of this WG's 100KB Whh slice into LDS ----
  // D# (ISA ch8): 3D tile: dim0 = 2048B contiguous (two ntG blocks), dim1 = 5 gates
  // (stride 20480B), dim2 = 10 k-tiles (stride 102400B). data_size = 8B units.
  if (wave == 0) {
    unsigned lds_off = (unsigned)(uintptr_t)(void*)wlds;
    unsigned long long ga =
        (unsigned long long)(uintptr_t)whhfrag + (unsigned long long)wg * 2048ull;
    v4u_t g0;
    g0[0] = 1u;                                            // count=1, user mode
    g0[1] = lds_off;                                       // lds_addr
    g0[2] = (unsigned)(ga & 0xffffffffull);                // global_addr[31:0]
    g0[3] = (unsigned)((ga >> 32) & 0x1ffffffull) | (2u << 30); // addr[56:32]|type=2
    const unsigned td0 = 256, td1 = 5, tile0 = 256, tile1 = 5, tile2 = 10;
    const unsigned long long s0 = 2560ull, s1 = 12800ull;  // strides in 8B units
    v8u_t g1;
    g1[0] = (3u << 16);                                    // data_size = 8 bytes
    g1[1] = (td0 & 0xffffu) << 16;                         // tensor_dim0[15:0]
    g1[2] = (td0 >> 16) | ((td1 & 0xffffu) << 16);         // td0[31:16] | td1[15:0]
    g1[3] = (td1 >> 16) | (tile0 << 16);                   // td1[31:16] | tile_dim0
    g1[4] = tile1 | (tile2 << 16);                         // tile_dim1 | tile_dim2
    g1[5] = (unsigned)(s0 & 0xffffffffull);                // dim0_stride[31:0]
    g1[6] = (unsigned)(s0 >> 32) | ((unsigned)(s1 & 0xffffull) << 16);
    g1[7] = (unsigned)((s1 >> 16) & 0xffffffffull);        // dim1_stride[47:16]
    v4u_t g2; g2[0] = 10u; g2[1] = 0u; g2[2] = 0u; g2[3] = 0u; // tensor_dim2 = 10
    v4u_t g3; g3[0] = 0u;  g3[1] = 0u; g3[2] = 0u; g3[3] = 0u;
    asm volatile("tensor_load_to_lds %0, %1, %2, %3\n\t"
                 "s_wait_tensorcnt 0x0"
                 :
                 : "s"(g0), "s"(g1), "s"(g2), "s"(g3)
                 : "memory");
  }
  __syncthreads();

  float cst[8], hst[8];
#pragma unroll
  for (int e = 0; e < 8; ++e) { cst[e] = 0.0f; hst[e] = 0.0f; }

  for (int s = 0; s < TT; ++s) {
    const int t = rev ? (TT - 1 - s) : s;
    const __bf16* hsrc = h_bf + (size_t)(s & 1) * BB * KP;
    __bf16* hdst = h_bf + (size_t)((s & 1) ^ 1) * BB * KP;

    if (tid < BB)
      __builtin_prefetch(gx + (size_t)(t * BB + tid) * NP6 + wg * 32, 0, 3);

    // gh GEMM: 40 tiles, 5/wave; A from global h (double-buffered), B from LDS
#pragma unroll
    for (int i = 0; i < 5; ++i) {
      int tIdx = wave * 5 + i;
      int mt = tIdx / 10, g = (tIdx % 10) >> 1, nt = tIdx & 1;
      int row = mt * 16 + (lane & 15);
      int kb  = (lane & 16) ? 8 : 0;
      v8f_t acc = v8f_t{};
      v16bf_t a = load_a_frag(hsrc + (size_t)row * KP + kb);
#pragma unroll
      for (int kt = 0; kt < KTN; ++kt) {
        v16bf_t an;
        if (kt < KTN - 1)
          an = load_a_frag(hsrc + (size_t)row * KP + (kt + 1) * 32 + kb);
        v16bf_t b = load_b_frag(wlds + ((size_t)((kt * 5 + g) * 2 + nt) * 512 + lane * 16));
        acc = wmma_bf16(a, b, acc);
        if (kt < KTN - 1) a = an;
      }
      int r0 = (lane & 16) ? 8 : 0;
      int col = nt * 16 + (lane & 15);
#pragma unroll
      for (int r = 0; r < 8; ++r)
        gh[(g * BB + mt * 16 + r + r0) * 32 + col] = acc[r];
    }
    __syncthreads();

    // elementwise cell update: 8 (b,c) slots per thread, state in registers
#pragma unroll
    for (int e = 0; e < 8; ++e) {
      int fl = e * 256 + tid;
      int b  = fl >> 5;
      int c  = fl & 31;
      int colp = wg * 32 + c;
      bool real = colp < HH;
      const float* gxr = gx + (size_t)(t * BB + b) * NP6;
      float bi[6];
#pragma unroll
      for (int g = 0; g < 6; ++g) bi[g] = real ? bias[g * HH + colp] : 0.0f;
      float ii = sigf(gxr[0 * KP + colp] + bi[0] + gh[(0 * BB + b) * 32 + c]);
      float ff = sigf(gxr[1 * KP + colp] + bi[1] + gh[(1 * BB + b) * 32 + c]);
      float oo = sigf(gxr[2 * KP + colp] + bi[2] + gh[(2 * BB + b) * 32 + c]);
      float gg = tanhf(gxr[3 * KP + colp] + bi[3] + gh[(3 * BB + b) * 32 + c]);
      float rr = sigf(gxr[4 * KP + colp] + bi[4] + gh[(4 * BB + b) * 32 + c]);
      float kk = gxr[5 * KP + colp] + bi[5];
      float cn = ff * cst[e] + ii * gg;
      float hn = rr * (oo * tanhf(cn)) + (1.0f - rr) * kk;
      float m  = masks[t * BB + b];
      hn = m * hn + (1.0f - m) * hst[e];
      cn = m * cn + (1.0f - m) * cst[e];
      hst[e] = hn; cst[e] = cn;
      if (real) seq_out[(size_t)(t * BB + b) * HH + colp] = hn;
      hdst[b * KP + colp] = (__bf16)hn;
    }

    // grid-wide step barrier (monotonic counter, reset by memsetAsync per launch)
    __syncthreads();
    if (tid == 0) {
      __hip_atomic_fetch_add(bar, 1u, __ATOMIC_RELEASE, __HIP_MEMORY_SCOPE_AGENT);
      unsigned target = (unsigned)(s + 1) * (unsigned)NWG;
      while (__hip_atomic_load(bar, __ATOMIC_ACQUIRE, __HIP_MEMORY_SCOPE_AGENT) < target)
        __builtin_amdgcn_s_sleep(1);
    }
    __syncthreads();
  }

  // final hT / cT -> d_out sections
  const size_t HOFF = (size_t)ROWS * HH;
  const size_t COFF = HOFF + (size_t)8 * BB * HH;
#pragma unroll
  for (int e = 0; e < 8; ++e) {
    int fl = e * 256 + tid;
    int b  = fl >> 5;
    int c  = fl & 31;
    int colp = wg * 32 + c;
    if (colp < HH) {
      hc_out[HOFF + (size_t)scanIdx * BB * HH + b * HH + colp] = hst[e];
      hc_out[COFF + (size_t)scanIdx * BB * HH + b * HH + colp] = cst[e];
    }
  }
}

extern "C" void kernel_launch(void* const* d_in, const int* in_sizes, int n_in,
                              void* d_out, int out_size, void* d_ws, size_t ws_size,
                              hipStream_t stream) {
  (void)in_sizes; (void)n_in; (void)out_size; (void)ws_size;
  const float* x     = (const float*)d_in[0];
  const float* masks = (const float*)d_in[1];
  const float* Wih_f = (const float*)d_in[2];
  const float* Whh_f = (const float*)d_in[3];
  const float* b_f   = (const float*)d_in[4];
  const float* Wih_b = (const float*)d_in[5];
  const float* Whh_b = (const float*)d_in[6];
  const float* b_b   = (const float*)d_in[7];
  float* out = (float*)d_out;

  char* ws = (char*)d_ws;
  size_t off = 0;
  float*  gx       = (float*)(ws + off);  off += (size_t)ROWS * NP6 * 4;
  __bf16* in_bf    = (__bf16*)(ws + off); off += (size_t)ROWS * KP * 2;
  float*  seq_out  = (float*)(ws + off);  off += (size_t)ROWS * HH * 4;
  __bf16* wih_frag = (__bf16*)(ws + off); off += (size_t)KP * NP6 * 2;
  __bf16* whh_frag = (__bf16*)(ws + off); off += (size_t)KP * 1600 * 2;
  __bf16* h_bf     = (__bf16*)(ws + off); off += (size_t)2 * BB * KP * 2;
  unsigned* bar    = (unsigned*)(ws + off); off += 256;

  const int nPackX  = ROWS * KP;
  const int nPackW6 = KTN * NT6 * 32 * 16;
  const int nPackW5 = KTN * NT5 * 32 * 16;

  for (int d = 0; d < 8; ++d) {
    const int layer = d >> 1;
    const int isB   = d & 1;
    const float* src = (d == 0) ? x : seq_out;
    const float* Wih = (isB ? Wih_b : Wih_f) + (size_t)layer * HH * 6 * HH;
    const float* Whh = (isB ? Whh_b : Whh_f) + (size_t)layer * HH * 5 * HH;
    const float* bb  = (isB ? b_b   : b_f)   + (size_t)layer * 6 * HH;

    pack_x_kernel<<<nPackX / 256, 256, 0, stream>>>(src, in_bf, nPackX);
    pack_w_kernel<<<nPackW6 / 256, 256, 0, stream>>>(Wih, wih_frag, 6, NT6, nPackW6);
    pack_w_kernel<<<nPackW5 / 256, 256, 0, stream>>>(Whh, whh_frag, 5, NT5, nPackW5);
    gemm_gx_kernel<<<3840, 256, 0, stream>>>(in_bf, wih_frag, gx);

    hipMemsetAsync(h_bf, 0, (size_t)2 * BB * KP * 2, stream);
    hipMemsetAsync(bar, 0, sizeof(unsigned), stream);

    float* sout = (d == 7) ? out : seq_out;
    scan_kernel<<<NWG, 256, 0, stream>>>(gx, whh_frag, bb, masks, sout, out,
                                         h_bf, bar, d, isB);
  }
}
